// RSSM_68650757259650
// MI455X (gfx1250) — matrix operand order repaired and checked by reading, not compiled, not run on hardware
//
#include <hip/hip_runtime.h>
#include <math.h>

typedef __attribute__((ext_vector_type(16))) _Float16 v16h;
typedef __attribute__((ext_vector_type(2)))  _Float16 h2;
typedef __attribute__((ext_vector_type(8)))  float    v8f;

#define B_    512
#define T_    256
#define LAT   30
#define ACT   6
#define ENCD  1024
#define HID   200

// ---------------------------------------------------------------------------
// Weight packing: f32 row-major W[N][K] (+ bias as augmented row k==K)
//   ->  f16 WMMA B-fragments for V_WMMA_F32_16X16X32_F16 (wave32):
//   lane = h*16 + n  (h = K-half, n = column)
//   dword j of the v16h holds K = ktile*32 + h*16 + 2j (+1)
// Packed as dst[tile][lane][16 f16], tile = nt*Kt + kt  (512 f16 / tile).
// ---------------------------------------------------------------------------
__global__ void pack_w(const float* __restrict__ A,  const float* __restrict__ A2,
                       const float* __restrict__ bA, const float* __restrict__ bB,
                       int rowsA, int N, int K, int Kt, _Float16* __restrict__ dst)
{
    int tile = blockIdx.x;
    int nt = tile / Kt, kt = tile - nt * Kt;
    for (int e = threadIdx.x; e < 512; e += blockDim.x) {
        int lane = e >> 4, s = e & 15;
        int n  = lane & 15, hh = lane >> 4;
        int j  = s >> 1,  par = s & 1;
        int k   = kt * 32 + hh * 16 + 2 * j + par;
        int col = nt * 16 + n;
        float v = 0.f;
        if (col < N) {
            if (k < K)
                v = (col < rowsA) ? A[(size_t)col * K + k]
                                  : A2[(size_t)(col - rowsA) * K + k];
            else if (k == K)                       // augmented bias row
                v = (col < rowsA) ? bA[col] : bB[col - rowsA];
        }
        dst[((size_t)tile << 9) + e] = (_Float16)v;
    }
}

// ---------------------------------------------------------------------------
// Straight-line WMMA accumulation chains (no branches, no register rotation).
// ---------------------------------------------------------------------------
template <int Kt>
__device__ __forceinline__ void tile_pair(const _Float16* __restrict__ bp0,
                                          const _Float16* __restrict__ bp1,
                                          const unsigned int* __restrict__ arow,
                                          int hh, v8f& c0, v8f& c1)
{
#pragma unroll 8
    for (int kt = 0; kt < Kt; ++kt) {
        union { unsigned int u[8]; v16h v; } au;
        const unsigned int* ar32 = arow + (kt << 4);
        *(uint4*)&au.u[0] = *(const uint4*)(ar32 + 8 * hh);
        *(uint4*)&au.u[4] = *(const uint4*)(ar32 + 8 * hh + 4);
        v16h b0 = *(const v16h*)(bp0 + ((size_t)kt << 9));
        v16h b1 = *(const v16h*)(bp1 + ((size_t)kt << 9));
        c0 = __builtin_amdgcn_wmma_f32_16x16x32_f16(
                 false, au.v, false, b0, (short)0, c0, false, false);
        c1 = __builtin_amdgcn_wmma_f32_16x16x32_f16(
                 false, au.v, false, b1, (short)0, c1, false, false);
    }
}

template <int Kt>
__device__ __forceinline__ void tile_single(const _Float16* __restrict__ bp0,
                                            const unsigned int* __restrict__ arow,
                                            int hh, v8f& c0)
{
#pragma unroll 8
    for (int kt = 0; kt < Kt; ++kt) {
        union { unsigned int u[8]; v16h v; } au;
        const unsigned int* ar32 = arow + (kt << 4);
        *(uint4*)&au.u[0] = *(const uint4*)(ar32 + 8 * hh);
        *(uint4*)&au.u[4] = *(const uint4*)(ar32 + 8 * hh + 4);
        v16h b0 = *(const v16h*)(bp0 + ((size_t)kt << 9));
        c0 = __builtin_amdgcn_wmma_f32_16x16x32_f16(
                 false, au.v, false, b0, (short)0, c0, false, false);
    }
}

// ---------------------------------------------------------------------------
// M=16 GEMM engine, dual n-tile per wave (nt, nt+8): two independent WMMA
// chains share one A fragment. `wave` is an SGPR (readfirstlane) and Kt/Nt
// are compile-time, so the dual/single decision is one scalar branch per
// n-tile pair, entirely outside the K loop.
// ---------------------------------------------------------------------------
template <int Kt, int Nt, typename EmitF>
__device__ __forceinline__ void gemm16(const _Float16* __restrict__ Bw,
                                       const _Float16* __restrict__ A,
                                       int Kpad, int wave, int lane, EmitF emit)
{
    const int row = lane & 15;      // batch row for A == output column n for C
    const int hh  = lane >> 4;
    const unsigned int* arow = (const unsigned int*)(A + row * Kpad);
    for (int nt0 = wave; nt0 < Nt; nt0 += 16) {
        const _Float16* bp0 = Bw + (((size_t)nt0 * Kt) << 9) + ((size_t)lane << 4);
        if (nt0 + 8 < Nt) {                        // scalar, once per pass
            const _Float16* bp1 = bp0 + (((size_t)8 * Kt) << 9);
            v8f c0 = {}, c1 = {};
            tile_pair<Kt>(bp0, bp1, arow, hh, c0, c1);
            emit(nt0 * 16 + row, hh, c0);
            emit((nt0 + 8) * 16 + row, hh, c1);
        } else {
            v8f c0 = {};
            tile_single<Kt>(bp0, arow, hh, c0);
            emit(nt0 * 16 + row, hh, c0);
        }
    }
}

// Packed-f16 ReLU emit into a stride-224 f16 buffer: convert first, then
// v_pk_max_num_f16 + dword-level live/fill select (col 200 = bias 1.0 row).
__device__ __forceinline__ void emit_relu224(_Float16* __restrict__ dst,
                                             int col, int hh, const v8f& c)
{
    const bool         live  = (col < 200);
    const unsigned int fillb = (col == 200) ? 0x3C003C00u : 0u;  // f16 {1,1}
    _Float16* p = dst + (hh << 3) * 224 + col;
#pragma unroll
    for (int m = 0; m < 4; ++m) {
        h2 v;
        v.x = (_Float16)c[2 * m];
        v.y = (_Float16)c[2 * m + 1];
        h2 z = {(_Float16)0.f, (_Float16)0.f};
        h2 r = __builtin_elementwise_max(v, z);
        unsigned int bits = live ? __builtin_bit_cast(unsigned int, r) : fillb;
        h2 o = __builtin_bit_cast(h2, bits);
        p[(2 * m) * 224]     = o.x;
        p[(2 * m + 1) * 224] = o.y;
    }
}

struct RssmArgs {
    const float *prev_s, *prev_h, *enc, *act, *epsp, *epsq, *b_hh;
    const _Float16 *w_sa1, *w_sa2, *w_ih, *wp1, *wp2, *wph,
                   *wq1, *wq2, *wqh, *wr1, *wr2, *wr3;
    float *o_prior, *o_post, *o_h, *o_mup, *o_lvp, *o_muq, *o_lvq, *o_rew;
};

__device__ __forceinline__ float sigmf(float x) { return 1.f / (1.f + __expf(-x)); }

// ---------------------------------------------------------------------------
// Main rollout kernel: 1 block = 16 batch rows, loops over all T steps.
// ---------------------------------------------------------------------------
__global__ __launch_bounds__(256) void rssm_kernel(RssmArgs g)
{
    __shared__ __align__(16) _Float16 s_in [16 * 64];    // [post|a|1|0] K=36+1
    __shared__ __align__(16) _Float16 s_sah[16 * 224];   // sa hidden  K=200+1
    __shared__ __align__(16) _Float16 s_x  [16 * 288];   // [sa|h|1|0] K=264+1
    __shared__ __align__(16) _Float16 s_h  [16 * 224];   // h_new      K=200+1
    __shared__ __align__(16) _Float16 s_gi [16 * 608];   // GRU pre-activations
    __shared__ __align__(16) _Float16 s_q  [16 * 1248];  // [h|enc|1|0] K=1224+1
    __shared__ __align__(16) _Float16 s_mA [16 * 224];   // MLP ping   K=200+1
    __shared__ __align__(16) _Float16 s_mB [16 * 224];   // MLP pong   K=200+1
    __shared__ __align__(16) _Float16 s_r  [16 * 256];   // [post|h|1|0] K=230+1
    __shared__ __align__(16) float    s_ml [16 * 64];    // head out: mu|lv
    __shared__ __align__(16) float    s_bhh[608];        // GRU hidden bias

    const int tid  = threadIdx.x;
    const int wave = __builtin_amdgcn_readfirstlane(tid >> 5);  // SGPR wave id
    const int lane = tid & 31;
    const int b0   = blockIdx.x * 16;

    // zero everything once (padded K columns must read as 0 forever after)
    for (int i = tid; i < 16 * 64;   i += 256) s_in[i]  = (_Float16)0.f;
    for (int i = tid; i < 16 * 224;  i += 256) s_sah[i] = (_Float16)0.f;
    for (int i = tid; i < 16 * 288;  i += 256) s_x[i]   = (_Float16)0.f;
    for (int i = tid; i < 16 * 224;  i += 256) s_h[i]   = (_Float16)0.f;
    for (int i = tid; i < 16 * 1248; i += 256) s_q[i]   = (_Float16)0.f;
    for (int i = tid; i < 16 * 224;  i += 256) s_mA[i]  = (_Float16)0.f;
    for (int i = tid; i < 16 * 224;  i += 256) s_mB[i]  = (_Float16)0.f;
    for (int i = tid; i < 16 * 256;  i += 256) s_r[i]   = (_Float16)0.f;
    for (int i = tid; i < 600;       i += 256) s_bhh[i] = g.b_hh[i];

    // augmented 1.0 columns (bias rows of consuming GEMMs); persist forever
    if (tid < 16) {
        s_in [tid * 64   + 36]   = (_Float16)1.f;
        s_x  [tid * 288  + 264]  = (_Float16)1.f;
        s_h  [tid * 224  + 200]  = (_Float16)1.f;
        s_q  [tid * 1248 + 1224] = (_Float16)1.f;
        s_r  [tid * 256  + 230]  = (_Float16)1.f;
    }

    // carry init: post = prev_stochastic_state, h = prev_hidden
    for (int e = tid; e < 16 * LAT; e += 256) {
        int r = e / LAT, i = e - r * LAT;
        s_in[r * 64 + i] = (_Float16)g.prev_s[(size_t)(b0 + r) * LAT + i];
    }
    for (int e = tid; e < 16 * HID; e += 256) {
        int r = e / HID, c = e - r * HID;
        s_x[r * 288 + 64 + c] = (_Float16)g.prev_h[(size_t)(b0 + r) * HID + c];
    }
    __syncthreads();

    for (int t = 0; t < T_; ++t) {
        // ---- stage per-step inputs (streaming: non-temporal) -----------
        for (int e = tid; e < 16 * ACT; e += 256) {
            int r = e / ACT, i = e - r * ACT;
            s_in[r * 64 + LAT + i] = (_Float16)__builtin_nontemporal_load(
                &g.act[((size_t)(b0 + r) * T_ + t) * ACT + i]);
        }
        for (int e = tid; e < 16 * ENCD; e += 256) {
            int r = e >> 10, k = e & 1023;
            s_q[r * 1248 + 200 + k] = (_Float16)__builtin_nontemporal_load(
                &g.enc[((size_t)(b0 + r) * T_ + t) * ENCD + k]);
        }
        __syncthreads();

        // ---- state-action MLP (bias folded into k=36 / k=200 rows) -----
        gemm16<2, 13>(g.w_sa1, s_in, 64, wave, lane,
               [&](int col, int hh, const v8f& c) { emit_relu224(s_sah, col, hh, c); });
        __syncthreads();
        gemm16<7, 4>(g.w_sa2, s_sah, 224, wave, lane,
               [&](int col, int hh, const v8f& c) {
#pragma unroll
                   for (int j = 0; j < 8; ++j)
                       s_x[(j + (hh << 3)) * 288 + col] = (_Float16)c[j];
               });
        __syncthreads();

        // ---- GRU input gemm: gi = W_ih @ [sa | h_prev] + b_ih ----------
        gemm16<9, 38>(g.w_ih, s_x, 288, wave, lane,
               [&](int col, int hh, const v8f& c) {
#pragma unroll
                   for (int j = 0; j < 8; ++j)
                       s_gi[(j + (hh << 3)) * 608 + col] = (_Float16)c[j];
               });

        // ---- prefetch next step's encoded obs into cache (no counters) --
        if (t + 1 < T_) {
#pragma unroll
            for (int p = 0; p < 2; ++p) {
                int L = tid + p * 256;               // 512 lines of 128 B
                int r = L >> 5, off = (L & 31) << 5; // offset in floats
                __builtin_prefetch(
                    g.enc + ((size_t)(b0 + r) * T_ + (t + 1)) * ENCD + off, 0, 1);
            }
        }
        __syncthreads();

        // ---- gates (h0 = 0, so gh = b_hh) ------------------------------
        for (int e = tid; e < 16 * HID; e += 256) {
            int r = e / HID, c = e - r * HID;
            float rg = sigmf((float)s_gi[r * 608 + c]       + s_bhh[c]);
            float zg = sigmf((float)s_gi[r * 608 + 200 + c] + s_bhh[200 + c]);
            float ng = tanhf((float)s_gi[r * 608 + 400 + c] + rg * s_bhh[400 + c]);
            float h  = (1.f - zg) * ng;
            __builtin_nontemporal_store(h, &g.o_h[((size_t)(b0 + r) * T_ + t) * HID + c]);
            _Float16 hf = (_Float16)h;
            s_h[r * 224 + c]      = hf;   // prior input
            s_q[r * 1248 + c]     = hf;   // posterior input (concat head)
            s_r[r * 256 + 30 + c] = hf;   // reward input
            s_x[r * 288 + 64 + c] = hf;   // carry to next step
        }
        __syncthreads();

        // ---- prior ------------------------------------------------------
        gemm16<7, 13>(g.wp1, s_h, 224, wave, lane,
               [&](int col, int hh, const v8f& c) { emit_relu224(s_mA, col, hh, c); });
        __syncthreads();
        gemm16<7, 13>(g.wp2, s_mA, 224, wave, lane,
               [&](int col, int hh, const v8f& c) { emit_relu224(s_mB, col, hh, c); });
        __syncthreads();
        gemm16<7, 4>(g.wph, s_mB, 224, wave, lane,
               [&](int col, int hh, const v8f& c) {
#pragma unroll
                   for (int j = 0; j < 8; ++j)
                       s_ml[(j + (hh << 3)) * 64 + col] = c[j];
               });
        __syncthreads();
        for (int e = tid; e < 16 * LAT; e += 256) {
            int r = e / LAT, i = e - r * LAT;
            float mu = s_ml[r * 64 + i], lv = s_ml[r * 64 + 30 + i];
            size_t oi = ((size_t)(b0 + r) * T_ + t) * LAT + i;
            float ep = __builtin_nontemporal_load(&g.epsp[oi]);
            __builtin_nontemporal_store(mu, &g.o_mup[oi]);
            __builtin_nontemporal_store(lv, &g.o_lvp[oi]);
            __builtin_nontemporal_store(mu + ep * __expf(0.5f * lv), &g.o_prior[oi]);
        }
        __syncthreads();

        // ---- posterior --------------------------------------------------
        gemm16<39, 13>(g.wq1, s_q, 1248, wave, lane,
               [&](int col, int hh, const v8f& c) { emit_relu224(s_mA, col, hh, c); });
        __syncthreads();
        gemm16<7, 13>(g.wq2, s_mA, 224, wave, lane,
               [&](int col, int hh, const v8f& c) { emit_relu224(s_mB, col, hh, c); });
        __syncthreads();
        gemm16<7, 4>(g.wqh, s_mB, 224, wave, lane,
               [&](int col, int hh, const v8f& c) {
#pragma unroll
                   for (int j = 0; j < 8; ++j)
                       s_ml[(j + (hh << 3)) * 64 + col] = c[j];
               });
        __syncthreads();
        for (int e = tid; e < 16 * LAT; e += 256) {
            int r = e / LAT, i = e - r * LAT;
            float mu = s_ml[r * 64 + i], lv = s_ml[r * 64 + 30 + i];
            size_t oi = ((size_t)(b0 + r) * T_ + t) * LAT + i;
            float eq = __builtin_nontemporal_load(&g.epsq[oi]);
            float ps = mu + eq * __expf(0.5f * lv);
            __builtin_nontemporal_store(mu, &g.o_muq[oi]);
            __builtin_nontemporal_store(lv, &g.o_lvq[oi]);
            __builtin_nontemporal_store(ps, &g.o_post[oi]);
            _Float16 pf = (_Float16)ps;
            s_r[r * 256 + i] = pf;   // reward input head
            s_in[r * 64 + i] = pf;   // carry post -> next step
        }
        __syncthreads();

        // ---- reward -----------------------------------------------------
        gemm16<8, 13>(g.wr1, s_r, 256, wave, lane,
               [&](int col, int hh, const v8f& c) { emit_relu224(s_mA, col, hh, c); });
        __syncthreads();
        gemm16<7, 13>(g.wr2, s_mA, 224, wave, lane,
               [&](int col, int hh, const v8f& c) { emit_relu224(s_mB, col, hh, c); });
        __syncthreads();
        gemm16<7, 1>(g.wr3, s_mB, 224, wave, lane,
               [&](int col, int hh, const v8f& c) {
                   if (col == 0) {
#pragma unroll
                       for (int j = 0; j < 8; ++j)
                           __builtin_nontemporal_store(
                               c[j], &g.o_rew[(size_t)(b0 + j + (hh << 3)) * T_ + t]);
                   }
               });
        __syncthreads();
    }
}

// ---------------------------------------------------------------------------
// Workspace layout (f16 elements): packed weight fragments.
// ---------------------------------------------------------------------------
#define OFF_SA1 0u
#define OFF_SA2 13312u
#define OFF_IH  27648u
#define OFF_P1  202752u
#define OFF_P2  249344u
#define OFF_PH  295936u
#define OFF_Q1  310272u
#define OFF_Q2  569856u
#define OFF_QH  616448u
#define OFF_R1  630784u
#define OFF_R2  684032u
#define OFF_R3  730624u    /* end = 734208 f16 = ~1.4 MB */

extern "C" void kernel_launch(void* const* d_in, const int* in_sizes, int n_in,
                              void* d_out, int out_size, void* d_ws, size_t ws_size,
                              hipStream_t stream)
{
    (void)in_sizes; (void)n_in; (void)out_size; (void)ws_size;
    // input order: 0 prev_s, 1 prev_h, 2 enc, 3 act, 4 eps_p, 5 eps_q, params:
    // 6 W_sa1,7 b_sa1,8 W_sa2,9 b_sa2,10 W_ih,11 b_ih,12 W_hh,13 b_hh,
    // 14 Wp1,15 bp1,16 Wp2,17 bp2,18 Wpmu,19 bpmu,20 Wplv,21 bplv,
    // 22 Wq1,23 bq1,24 Wq2,25 bq2,26 Wqmu,27 bqmu,28 Wqlv,29 bqlv,
    // 30 Wr1,31 br1,32 Wr2,33 br2,34 Wr3,35 br3
#define IN(i) ((const float*)d_in[(i)])
    _Float16* ws = (_Float16*)d_ws;

    // ---- pack all weights (+bias rows) to f16 WMMA fragments ----
    pack_w<<< 26, 256, 0, stream>>>(IN(6),  IN(6),  IN(7),  IN(7),  200, 200,   36,  2, ws + OFF_SA1);
    pack_w<<< 28, 256, 0, stream>>>(IN(8),  IN(8),  IN(9),  IN(9),   64,  64,  200,  7, ws + OFF_SA2);
    pack_w<<<342, 256, 0, stream>>>(IN(10), IN(10), IN(11), IN(11), 600, 600,  264,  9, ws + OFF_IH);
    pack_w<<< 91, 256, 0, stream>>>(IN(14), IN(14), IN(15), IN(15), 200, 200,  200,  7, ws + OFF_P1);
    pack_w<<< 91, 256, 0, stream>>>(IN(16), IN(16), IN(17), IN(17), 200, 200,  200,  7, ws + OFF_P2);
    pack_w<<< 28, 256, 0, stream>>>(IN(18), IN(20), IN(19), IN(21),  30,  60,  200,  7, ws + OFF_PH);
    pack_w<<<507, 256, 0, stream>>>(IN(22), IN(22), IN(23), IN(23), 200, 200, 1224, 39, ws + OFF_Q1);
    pack_w<<< 91, 256, 0, stream>>>(IN(24), IN(24), IN(25), IN(25), 200, 200,  200,  7, ws + OFF_Q2);
    pack_w<<< 28, 256, 0, stream>>>(IN(26), IN(28), IN(27), IN(29),  30,  60,  200,  7, ws + OFF_QH);
    pack_w<<<104, 256, 0, stream>>>(IN(30), IN(30), IN(31), IN(31), 200, 200,  230,  8, ws + OFF_R1);
    pack_w<<< 91, 256, 0, stream>>>(IN(32), IN(32), IN(33), IN(33), 200, 200,  200,  7, ws + OFF_R2);
    pack_w<<<  7, 256, 0, stream>>>(IN(34), IN(34), IN(35), IN(35),   1,   1,  200,  7, ws + OFF_R3);

    // ---- output slices (return order) ----
    float* out = (float*)d_out;
    const size_t S30 = (size_t)B_ * T_ * LAT, S200 = (size_t)B_ * T_ * HID;

    RssmArgs a;
    a.prev_s = IN(0); a.prev_h = IN(1); a.enc = IN(2); a.act = IN(3);
    a.epsp = IN(4); a.epsq = IN(5); a.b_hh = IN(13);
    a.w_sa1 = ws + OFF_SA1; a.w_sa2 = ws + OFF_SA2; a.w_ih = ws + OFF_IH;
    a.wp1 = ws + OFF_P1; a.wp2 = ws + OFF_P2; a.wph = ws + OFF_PH;
    a.wq1 = ws + OFF_Q1; a.wq2 = ws + OFF_Q2; a.wqh = ws + OFF_QH;
    a.wr1 = ws + OFF_R1; a.wr2 = ws + OFF_R2; a.wr3 = ws + OFF_R3;
    a.o_prior = out;
    a.o_post  = out + S30;
    a.o_h     = out + 2 * S30;
    a.o_mup   = out + 2 * S30 + S200;
    a.o_lvp   = a.o_mup + S30;
    a.o_muq   = a.o_lvp + S30;
    a.o_lvq   = a.o_muq + S30;
    a.o_rew   = a.o_lvq + S30;

    rssm_kernel<<<B_ / 16, 256, 0, stream>>>(a);
#undef IN
}
